// SimpleSubGMN_11699490914440
// MI455X (gfx1250) — compile-verified, matrix operand order checked by reading, not run on hardware
//
#include <hip/hip_runtime.h>
#include <hip/hip_bf16.h>

#define N_T 32768
#define N_Q 4096
#define E_T 1048576
#define E_Q 131072
#define H 128
#define NUM_LAYERS 3

typedef __attribute__((ext_vector_type(16))) _Float16 v16h;
typedef __attribute__((ext_vector_type(8)))  _Float16 v8h;
typedef __attribute__((ext_vector_type(8)))  float    v8f;

// ---------------------------------------------------------------------------
// WMMA operand loader (CDNA5 16-bit A/B layout, wave32):
//  lane l -> m/n = l&15, h = l>>4
//  v16h elements j=0..7  hold K = 32*kk + 8*h + j
//  v16h elements j=8..15 hold K = 32*kk + 16 + 8*h + (j-8)
// Both chunks are 8 contiguous halves (16B) -> b128 loads (global or ds).
// ---------------------------------------------------------------------------
__device__ __forceinline__ v16h load_tile_f16(const _Float16* row, int kk, int h) {
    const _Float16* p0 = row + 32 * kk + 8 * h;
    v8h lo = *(const v8h*)p0;
    v8h hi = *(const v8h*)(p0 + 16);
    v16h r;
#pragma unroll
    for (int j = 0; j < 8; ++j) { r[j] = lo[j]; r[8 + j] = hi[j]; }
    return r;
}

// ---------------------------------------------------------------------------
// 1) Embedding gather: x[n][h] = emb[ids[n]][h]
// ---------------------------------------------------------------------------
__global__ void k_gather(const int* __restrict__ ids, const float* __restrict__ emb,
                         float* __restrict__ x, int n) {
    int i = blockIdx.x * blockDim.x + threadIdx.x;
    if (i >= n * H) return;
    x[i] = emb[ids[i >> 7] * H + (i & 127)];
}

// ---------------------------------------------------------------------------
// 2) Edge scatter: agg[dst] += x[src]; cnt[dst] += 1.  32 lanes per edge.
// ---------------------------------------------------------------------------
__global__ void k_scatter(const int* __restrict__ ei, int E,
                          const float* __restrict__ x,
                          float* __restrict__ agg, float* __restrict__ cnt) {
    int t = blockIdx.x * blockDim.x + threadIdx.x;
    int e = t >> 5;
    if (e >= E) return;
    int lane = t & 31;
    int src = ei[e];
    int dst = ei[E + e];
    const float* xs = x + (size_t)src * H;
    float* ad = agg + (size_t)dst * H;
#pragma unroll
    for (int j = 0; j < 4; ++j)
        atomicAdd(ad + lane * 4 + j, xs[lane * 4 + j]);
    if (lane == 0) atomicAdd(cnt + dst, 1.0f);
}

// ---------------------------------------------------------------------------
// 3) SAGE layer (one wave per 16-row block, in place):
//    x[16 rows] = ELU( mean @ Wl^T + bl + x @ Wr^T )
//    A tiles staged to LDS as f16; B rows from pre-converted f16 W (L2-resident).
// ---------------------------------------------------------------------------
__global__ __launch_bounds__(32)
void k_sage(float* __restrict__ x, const float* __restrict__ agg,
            const float* __restrict__ cnt,
            const _Float16* __restrict__ Wl, const float* __restrict__ bl,
            const _Float16* __restrict__ Wr) {
    __shared__ __align__(16) _Float16 sm[16 * H];
    __shared__ __align__(16) _Float16 sx[16 * H];
    const int lane = threadIdx.x;
    const int row0 = blockIdx.x * 16;

    for (int i = lane; i < 16 * H; i += 32) {
        int r = i >> 7, c = i & 127;
        float inv = 1.0f / fmaxf(cnt[row0 + r], 1.0f);
        sm[i] = (_Float16)(agg[(size_t)(row0 + r) * H + c] * inv);
        sx[i] = (_Float16)x[(size_t)(row0 + r) * H + c];
    }
    __syncthreads();

    const int m = lane & 15, h = lane >> 4;
    const _Float16* srm = sm + m * H;
    const _Float16* srx = sx + m * H;

    for (int nt = 0; nt < 8; ++nt) {
        int col = nt * 16 + m;                 // output col == W row
        const _Float16* wl_row = Wl + (size_t)col * H;
        const _Float16* wr_row = Wr + (size_t)col * H;
        v8f acc = {};
#pragma unroll
        for (int kk = 0; kk < 4; ++kk) {
            v16h am = load_tile_f16(srm, kk, h);
            v16h bw = load_tile_f16(wl_row, kk, h);
            acc = __builtin_amdgcn_wmma_f32_16x16x32_f16(false, am, false, bw,
                                                         (short)0, acc, false, false);
            v16h ax = load_tile_f16(srx, kk, h);
            v16h br = load_tile_f16(wr_row, kk, h);
            acc = __builtin_amdgcn_wmma_f32_16x16x32_f16(false, ax, false, br,
                                                         (short)0, acc, false, false);
        }
        float bias = bl[col];
#pragma unroll
        for (int r = 0; r < 8; ++r) {
            float v = acc[r] + bias;
            v = v > 0.0f ? v : expm1f(v);       // ELU(alpha=1)
            x[(size_t)(row0 + r + 8 * h) * H + col] = v;
        }
    }
}

// ---------------------------------------------------------------------------
// 4) f32 -> f16 convert
// ---------------------------------------------------------------------------
__global__ void k_tohalf(const float* __restrict__ x, _Float16* __restrict__ y, int n) {
    int i = blockIdx.x * blockDim.x + threadIdx.x;
    if (i < n) y[i] = (_Float16)x[i];
}

// ---------------------------------------------------------------------------
// 5) Attention logits: out[q][t] = mask ? (eq[q].et[t])/sqrt(H) : -1e9
//    One wave handles 16 q-rows x 128 t-cols.  The 4KB contiguous eq tile is
//    staged to LDS via GLOBAL_LOAD_ASYNC_TO_LDS_B128 (ASYNCcnt), then fenced.
// ---------------------------------------------------------------------------
__global__ __launch_bounds__(32)
void k_attn(const _Float16* __restrict__ eq_h, const _Float16* __restrict__ et_h,
            const unsigned char* __restrict__ mask, float* __restrict__ out) {
    __shared__ __align__(16) _Float16 sq[16 * H];
    const int lane = threadIdx.x;
    const int q0 = blockIdx.x * 16;
    const int c0 = blockIdx.y * 128;

    // async stage: eq_h[q0*128 .. q0*128+2048) -> sq (contiguous 4KB copy)
    {
        const _Float16* gsrc = eq_h + (size_t)q0 * H;
        unsigned ldsbase = (unsigned)(unsigned long long)(&sq[0]);  // addr[31:0] = LDS byte addr
#pragma unroll
        for (int it = 0; it < 8; ++it) {
            unsigned lo = ldsbase + (unsigned)((it * 32 + lane) * 16);
            const void* ga = (const void*)(gsrc + (it * 32 + lane) * 8);
            asm volatile("global_load_async_to_lds_b128 %0, %1, off"
                         :: "v"(lo), "v"(ga) : "memory");
        }
        asm volatile("s_wait_asynccnt 0x0" ::: "memory");
    }
    __syncthreads();

    const int m = lane & 15, h = lane >> 4;
    const _Float16* srow = sq + m * H;
    const float scale = 0.08838834764831845f;  // 1/sqrt(128)

    for (int nt = 0; nt < 8; ++nt) {
        int col = c0 + nt * 16 + m;
        const _Float16* trow = et_h + (size_t)col * H;
        // prefetch next tile's B row
        if (nt < 7) __builtin_prefetch(trow + 16 * H, 0, 0);
        v8f acc = {};
#pragma unroll
        for (int kk = 0; kk < 4; ++kk) {
            v16h a = load_tile_f16(srow, kk, h);
            v16h b = load_tile_f16(trow, kk, h);
            acc = __builtin_amdgcn_wmma_f32_16x16x32_f16(false, a, false, b,
                                                         (short)0, acc, false, false);
        }
#pragma unroll
        for (int r = 0; r < 8; ++r) {
            size_t row = (size_t)(q0 + r + 8 * h);
            size_t idx = row * (size_t)N_T + (size_t)col;
            out[idx] = mask[idx] ? acc[r] * scale : -1e9f;
        }
    }
}

// ---------------------------------------------------------------------------
// 6) Per-row max + sum(exp(v-max)) (one block per q row)
// ---------------------------------------------------------------------------
__global__ __launch_bounds__(256)
void k_rowstats(const float* __restrict__ out, float* __restrict__ stats) {
    __shared__ float red[256];
    int row = blockIdx.x;
    const float* p = out + (size_t)row * N_T;
    float mx = -3.4e38f;
    for (int c = threadIdx.x; c < N_T; c += 256) mx = fmaxf(mx, p[c]);
    red[threadIdx.x] = mx; __syncthreads();
    for (int s = 128; s > 0; s >>= 1) {
        if (threadIdx.x < s) red[threadIdx.x] = fmaxf(red[threadIdx.x], red[threadIdx.x + s]);
        __syncthreads();
    }
    float rmax = red[0]; __syncthreads();
    float sum = 0.0f;
    for (int c = threadIdx.x; c < N_T; c += 256) sum += expf(p[c] - rmax);
    red[threadIdx.x] = sum; __syncthreads();
    for (int s = 128; s > 0; s >>= 1) {
        if (threadIdx.x < s) red[threadIdx.x] += red[threadIdx.x + s];
        __syncthreads();
    }
    if (threadIdx.x == 0) { stats[row * 2] = rmax; stats[row * 2 + 1] = red[0]; }
}

// ---------------------------------------------------------------------------
// 7) Normalize: out = exp(v - max)/sum
// ---------------------------------------------------------------------------
__global__ void k_norm(float* __restrict__ out, const float* __restrict__ stats) {
    size_t i = (size_t)blockIdx.x * blockDim.x + threadIdx.x;
    int row = (int)(i >> 15);                  // / N_T
    float v = out[i];
    out[i] = expf(v - stats[row * 2]) / stats[row * 2 + 1];
}

// ---------------------------------------------------------------------------
extern "C" void kernel_launch(void* const* d_in, const int* in_sizes, int n_in,
                              void* d_out, int out_size, void* d_ws, size_t ws_size,
                              hipStream_t stream) {
    const int*           target_x = (const int*)d_in[0];
    const int*           query_x  = (const int*)d_in[1];
    const int*           t_ei     = (const int*)d_in[2];
    const int*           q_ei     = (const int*)d_in[3];
    const unsigned char* mask     = (const unsigned char*)d_in[4];
    const float*         emb      = (const float*)d_in[5];
    const float*         Wl       = (const float*)d_in[6];
    const float*         bl       = (const float*)d_in[7];
    const float*         Wr       = (const float*)d_in[8];
    float* out = (float*)d_out;

    // workspace layout (256B aligned)
    char* ws = (char*)d_ws;
    size_t o = 0;
    auto alloc = [&](size_t bytes) { char* p = ws + o; o = (o + bytes + 255) & ~(size_t)255; return p; };
    float*    et_x   = (float*)alloc((size_t)N_T * H * 4);
    float*    et_agg = (float*)alloc((size_t)N_T * H * 4);
    float*    et_cnt = (float*)alloc((size_t)N_T * 4);
    float*    eq_x   = (float*)alloc((size_t)N_Q * H * 4);
    float*    eq_agg = (float*)alloc((size_t)N_Q * H * 4);
    float*    eq_cnt = (float*)alloc((size_t)N_Q * 4);
    _Float16* et_h   = (_Float16*)alloc((size_t)N_T * H * 2);
    _Float16* eq_h   = (_Float16*)alloc((size_t)N_Q * H * 2);
    _Float16* Wl_h   = (_Float16*)alloc((size_t)NUM_LAYERS * H * H * 2);
    _Float16* Wr_h   = (_Float16*)alloc((size_t)NUM_LAYERS * H * H * 2);
    float*    stats  = (float*)alloc((size_t)N_Q * 2 * 4);

    // 0) pre-convert weights to f16 (tiny, L2-resident thereafter)
    const int WN = NUM_LAYERS * H * H;
    k_tohalf<<<(WN + 255) / 256, 256, 0, stream>>>(Wl, Wl_h, WN);
    k_tohalf<<<(WN + 255) / 256, 256, 0, stream>>>(Wr, Wr_h, WN);

    // 1) embedding gather
    k_gather<<<(N_T * H + 255) / 256, 256, 0, stream>>>(target_x, emb, et_x, N_T);
    k_gather<<<(N_Q * H + 255) / 256, 256, 0, stream>>>(query_x, emb, eq_x, N_Q);

    // 2) three SAGE layers on both graphs
    for (int L = 0; L < NUM_LAYERS; ++L) {
        const _Float16* WlL = Wl_h + (size_t)L * H * H;
        const float*    blL = bl + (size_t)L * H;
        const _Float16* WrL = Wr_h + (size_t)L * H * H;

        hipMemsetAsync(et_agg, 0, (size_t)N_T * H * 4, stream);
        hipMemsetAsync(et_cnt, 0, (size_t)N_T * 4, stream);
        hipMemsetAsync(eq_agg, 0, (size_t)N_Q * H * 4, stream);
        hipMemsetAsync(eq_cnt, 0, (size_t)N_Q * 4, stream);

        k_scatter<<<(E_T * 32) / 256, 256, 0, stream>>>(t_ei, E_T, et_x, et_agg, et_cnt);
        k_scatter<<<(E_Q * 32) / 256, 256, 0, stream>>>(q_ei, E_Q, eq_x, eq_agg, eq_cnt);

        k_sage<<<N_T / 16, 32, 0, stream>>>(et_x, et_agg, et_cnt, WlL, blL, WrL);
        k_sage<<<N_Q / 16, 32, 0, stream>>>(eq_x, eq_agg, eq_cnt, WlL, blL, WrL);
    }

    // 3) attention + softmax
    k_tohalf<<<(N_T * H + 255) / 256, 256, 0, stream>>>(et_x, et_h, N_T * H);
    k_tohalf<<<(N_Q * H + 255) / 256, 256, 0, stream>>>(eq_x, eq_h, N_Q * H);

    dim3 ag(N_Q / 16, N_T / 128);
    k_attn<<<ag, 32, 0, stream>>>(eq_h, et_h, mask, out);

    k_rowstats<<<N_Q, 256, 0, stream>>>(out, stats);

    size_t total = (size_t)N_Q * N_T;
    k_norm<<<(unsigned)(total / 256), 256, 0, stream>>>(out, stats);
}
